// InferenceBox_49821620633907
// MI455X (gfx1250) — compile-verified, matrix operand order checked by reading, not run on hardware
//
#include <hip/hip_runtime.h>
#include <hip/hip_bf16.h>
#include <stdint.h>

#define BATCH 32
#define NBOX  25000
#define DPRED 85
#define NCONF 81          // softmax classes
#define NCLS  80          // output classes (drop background)
#define TOPK  200
#define NPAD  25088       // NBOX rounded up to multiple of 256
#define CONF_THR 0.01f
#define IOU_THR  0.45f
#define VAR_CENTER 0.1f
#define VAR_SIZE   0.2f

typedef __attribute__((ext_vector_type(16))) _Float16 v16h;
typedef __attribute__((ext_vector_type(8)))  float    v8f;
typedef __attribute__((ext_vector_type(4)))  unsigned v4u;
typedef __attribute__((ext_vector_type(8)))  int      v8i;
typedef __attribute__((ext_vector_type(4)))  int      v4i;

// ---------------------------------------------------------------------------
// TDM: async DMA of a 2D tile of f32 (rows x rowElems, row stride in elems)
// from global memory into LDS. Builds a D# per CDNA5 ISA:
//   group0: count=1 | lds_addr | global_addr[56:0] | type=2
//   group1: data_size=2 (4B), tensor_dim0/1, tile_dim0/1, dim0 stride
// This toolchain exposes the 6-arg builtin:
//   (uint32x4 g0, int32x8 g1, int32x4 g2, int32x4 g3, int32x8 extra, i32 cpol)
// ---------------------------------------------------------------------------
__device__ __forceinline__ void tdm_load_2d_f32(unsigned ldsByteAddr,
                                                const float* gptr,
                                                unsigned rows,
                                                unsigned rowElems,
                                                unsigned rowStrideElems) {
  unsigned long long ga = (unsigned long long)(uintptr_t)gptr;
  v4u g0;
  g0.x = 1u;                                                   // count=1, user desc
  g0.y = ldsByteAddr;                                          // lds_addr (bytes)
  g0.z = (unsigned)(ga & 0xFFFFFFFFull);                       // global_addr[31:0]
  g0.w = (unsigned)((ga >> 32) & 0x01FFFFFFull) | (2u << 30);  // addr[56:32] | type=2
  v8i g1;
  g1[0] = (int)(2u << 16);                                     // data_size = 4 bytes
  g1[1] = (int)((rowElems & 0xFFFFu) << 16);                   // tensor_dim0[15:0]
  g1[2] = (int)((rowElems >> 16) | ((rows & 0xFFFFu) << 16));  // dim0[31:16]|dim1[15:0]
  g1[3] = (int)((rows >> 16) | ((rowElems & 0xFFFFu) << 16));  // dim1[31:16]|tile_dim0
  g1[4] = (int)(rows & 0xFFFFu);                               // tile_dim1 | tile_dim2=0
  g1[5] = (int)rowStrideElems;                                 // dim0_stride[31:0]
  g1[6] = 0;
  g1[7] = 0;
  v4i z4 = {0, 0, 0, 0};                                       // groups 2/3 unused (2D)
  v8i z8 = {0, 0, 0, 0, 0, 0, 0, 0};
  __builtin_amdgcn_tensor_load_to_lds(g0, g1, z4, z4, z8, 0);
}

// ---------------------------------------------------------------------------
// Phase 1: one streaming pass over predicts (272 MB, HBM-bound ~12us floor).
// Per 256-thread block: 128 rows of one batch. Each wave TDM-loads its 16 rows
// (16 x 340B) into LDS, computes per-row softmax max and sum(exp) (sum via
// v_wmma_f32_16x16x32_f16 against an all-ones B matrix: one WMMA reduces 32
// classes for all 16 rows), and threads 0..127 decode boxes.
// A-matrix 16-bit layout (ISA 7.12.2): lane l (l<16) holds row l, K in
// {c0+0..7, c0+16..23}; lane l+16 holds K in {c0+8..15, c0+24..31}.
// ---------------------------------------------------------------------------
__global__ __launch_bounds__(256) void phase1_stats_decode(
    const float* __restrict__ predicts, const float* __restrict__ dboxes,
    float* __restrict__ locWS, float* __restrict__ rowMaxWS,
    float* __restrict__ invSumWS) {
  __shared__ float sPred[128 * DPRED];   // 43,520 B
  __shared__ float sSum[128];

  const int b   = blockIdx.y;
  const int r0  = blockIdx.x * 128;
  const int tid = threadIdx.x;
  const int wid  = tid >> 5;
  const int lane = tid & 31;
  const int l16  = lane & 15;
  const bool hi  = lane >= 16;

  // Per-wave async tensor DMA of its 16 rows, then wait on TENSORcnt.
  const int wr0 = r0 + wid * 16;
  int nrows = NBOX - wr0;
  if (nrows > 16) nrows = 16;
  if (nrows > 0) {
    unsigned ldsAddr = (unsigned)(uintptr_t)(&sPred[wid * 16 * DPRED]);
    tdm_load_2d_f32(ldsAddr, predicts + ((size_t)b * NBOX + wr0) * DPRED,
                    (unsigned)nrows, DPRED, DPRED);
    __builtin_amdgcn_s_wait_tensorcnt(0);
  }
  __syncthreads();

  const float* wrow = &sPred[wid * 16 * DPRED + l16 * DPRED];

  // Row max over this lane's K-subset, combined with partner lane (xor 16).
  float mx = -3.0e38f;
#pragma unroll
  for (int ch = 0; ch < 3; ++ch) {
#pragma unroll
    for (int e = 0; e < 16; ++e) {
      int k = ch * 32 + (e < 8 ? e : e + 8) + (hi ? 8 : 0);
      if (k < NCONF) mx = fmaxf(mx, wrow[4 + k]);
    }
  }
  mx = fmaxf(mx, __shfl_xor(mx, 16, 32));

  // sum(exp) via WMMA with B = ones: D[m][*] = sum_K exp(row m).
  v16h bOnes;
#pragma unroll
  for (int e = 0; e < 16; ++e) bOnes[e] = (_Float16)1.0f;
  v8f acc = {};
#pragma unroll
  for (int ch = 0; ch < 3; ++ch) {
    v16h a;
#pragma unroll
    for (int e = 0; e < 16; ++e) {
      int k = ch * 32 + (e < 8 ? e : e + 8) + (hi ? 8 : 0);
      float v = (k < NCONF) ? expf(wrow[4 + k] - mx) : 0.0f;
      a[e] = (_Float16)v;
    }
    acc = __builtin_amdgcn_wmma_f32_16x16x32_f16(false, a, false, bOnes,
                                                 (short)0, acc, false, false);
  }
  // D layout: VGPR r -> M=r (lanes 0-15), M=r+8 (lanes 16-31). Scatter sums.
#pragma unroll
  for (int r = 0; r < 8; ++r) {
    if (lane == r)      sSum[wid * 16 + r]     = acc[r];
    if (lane == 16 + r) sSum[wid * 16 + 8 + r] = acc[r];
  }
  __syncthreads();

  if (!hi) {
    int n = wr0 + l16;
    if (n < NBOX) {
      rowMaxWS[(size_t)b * NBOX + n] = mx;
      invSumWS[(size_t)b * NBOX + n] = 1.0f / sSum[wid * 16 + l16];
    }
  }

  // Box decode (threads 0..127, one row each) from the LDS-staged rows.
  if (tid < 128) {
    int n = r0 + tid;
    if (n < NBOX) {
      const float* p = &sPred[tid * DPRED];
      float dx = dboxes[n * 4 + 0], dy = dboxes[n * 4 + 1];
      float dw = dboxes[n * 4 + 2], dh = dboxes[n * 4 + 3];
      float cx = dx + p[0] * VAR_CENTER * dw;
      float cy = dy + p[1] * VAR_CENTER * dh;
      float w  = dw * expf(p[2] * VAR_SIZE);
      float h  = dh * expf(p[3] * VAR_SIZE);
      size_t o = ((size_t)b * NBOX + n) * 4;
      locWS[o + 0] = cx; locWS[o + 1] = cy;
      locWS[o + 2] = w;  locWS[o + 3] = h;
    }
  }
}

// ---------------------------------------------------------------------------
// Phase 2: one 256-thread workgroup per (b, c). All 25000 conf keys cached in
// LDS (100 KB — only possible with CDNA5's 320 KB/WGP). Exact 4-pass radix
// select of the 200th-largest, ordered compaction (ballot/popc, wave32),
// bitonic sort (value desc, index asc — matches lax.top_k ties), then NMS.
// ---------------------------------------------------------------------------
__global__ __launch_bounds__(256) void phase2_topk_nms(
    const float* __restrict__ predicts, const float* __restrict__ locWS,
    const float* __restrict__ rowMaxWS, const float* __restrict__ invSumWS,
    float* __restrict__ outBoxes, float* __restrict__ outKeep) {
  extern __shared__ unsigned char dynSmem[];
  unsigned* sKey    = (unsigned*)dynSmem;   // NPAD
  unsigned* sHist   = sKey + NPAD;          // 256
  unsigned* sTopKey = sHist + 256;          // 256
  unsigned* sTopIdx = sTopKey + 256;        // 256
  float* sX1 = (float*)(sTopIdx + 256);     // 256
  float* sY1 = sX1 + 256;
  float* sX2 = sY1 + 256;
  float* sY2 = sX2 + 256;
  float* sAr = sY2 + 256;
  int* sKeep = (int*)(sAr + 256);           // 256
  int* sWG   = sKeep + 256;                 // 8
  int* sWT   = sWG + 8;                     // 8
  int* sCtl  = sWT + 8;                     // 4: sel, kRemain, runG, runT

  const int c = blockIdx.x;
  const int b = blockIdx.y;
  const int tid = threadIdx.x;
  const size_t bn = (size_t)b * NBOX;
  const float* pb = predicts + bn * DPRED;

  // Build monotonic uint keys for masked conf (conf>thr ? conf : -1.0).
  for (int n = tid; n < NPAD; n += 256) {
    unsigned key = 0u;  // padding: ranks below everything
    if (n < NBOX) {
      float logit = pb[(size_t)n * DPRED + 4 + c];
      float conf = expf(logit - rowMaxWS[bn + n]) * invSumWS[bn + n];
      float v = (conf > CONF_THR) ? conf : -1.0f;
      unsigned u = __float_as_uint(v);
      key = (u & 0x80000000u) ? ~u : (u | 0x80000000u);
    }
    sKey[n] = key;
  }
  if (tid < 4) sCtl[tid] = 0;
  sTopKey[tid] = 0u;
  sTopIdx[tid] = 0x7FFFFFFFu;
  __syncthreads();

  // Exact radix select: key of the TOPK-th largest element.
  unsigned kth = 0u;
  int kRemain = TOPK;
  for (int p = 0; p < 4; ++p) {
    int shift = 24 - 8 * p;
    sHist[tid] = 0u;
    __syncthreads();
    for (int n = tid; n < NPAD; n += 256) {
      unsigned k = sKey[n];
      if (p == 0 || (k >> (shift + 8)) == (kth >> (shift + 8)))
        atomicAdd(&sHist[(k >> shift) & 0xFFu], 1u);
    }
    __syncthreads();
    if (tid == 0) {
      unsigned cum = 0; int sel = 0; int rem = kRemain;
      for (int bin = 255; bin >= 0; --bin) {
        unsigned hcnt = sHist[bin];
        if (cum + hcnt >= (unsigned)rem) { sel = bin; rem -= (int)cum; break; }
        cum += hcnt;
      }
      sCtl[0] = sel; sCtl[1] = rem;
    }
    __syncthreads();
    kth |= ((unsigned)sCtl[0]) << shift;
    kRemain = sCtl[1];
    __syncthreads();
  }
  const int nGreater = TOPK - kRemain;  // strictly-greater count

  // Ordered compaction: all keys > kth, plus first kRemain ties by index.
  const unsigned lane = tid & 31u;
  const unsigned wv = tid >> 5;
  const unsigned long long ltMask = (1ull << lane) - 1ull;
  for (int base = 0; base < NPAD; base += 256) {
    int n = base + tid;
    unsigned k = sKey[n];
    bool isG = (k > kth);
    bool isT = (k == kth);
    unsigned long long mG = __ballot(isG);
    unsigned long long mT = __ballot(isT);
    int offG = __popcll(mG & ltMask);
    int offT = __popcll(mT & ltMask);
    if (lane == 0) { sWG[wv] = __popcll(mG); sWT[wv] = __popcll(mT); }
    __syncthreads();
    if (tid == 0) {
      int rg = sCtl[2], rt = sCtl[3];
      for (int w = 0; w < 8; ++w) {
        int g = sWG[w]; sWG[w] = rg; rg += g;
        int t = sWT[w]; sWT[w] = rt; rt += t;
      }
      sCtl[2] = rg; sCtl[3] = rt;
    }
    __syncthreads();
    if (isG) {
      int slot = sWG[wv] + offG;               // < nGreater by construction
      sTopKey[slot] = k; sTopIdx[slot] = (unsigned)n;
    } else if (isT) {
      int t = sWT[wv] + offT;
      if (t < kRemain) {
        int slot = nGreater + t;
        sTopKey[slot] = k; sTopIdx[slot] = (unsigned)n;
      }
    }
    __syncthreads();
  }

  // Bitonic sort of 256 (200 real + pads): key desc, index asc on ties.
  for (unsigned sz = 2; sz <= 256; sz <<= 1) {
    for (unsigned st = sz >> 1; st > 0; st >>= 1) {
      __syncthreads();
      unsigned i = (unsigned)tid, j = i ^ st;
      if (j > i) {
        unsigned ka = sTopKey[i], kb = sTopKey[j];
        unsigned ia = sTopIdx[i], ib = sTopIdx[j];
        bool aFirst = (ka > kb) || (ka == kb && ia < ib);
        bool up = ((i & sz) == 0);
        if (aFirst != up) {
          sTopKey[i] = kb; sTopKey[j] = ka;
          sTopIdx[i] = ib; sTopIdx[j] = ia;
        }
      }
    }
  }
  __syncthreads();

  // Fetch boxes, init keep mask.
  float cx = 0.f, cy = 0.f, w = 0.f, h = 0.f;
  if (tid < TOPK) {
    unsigned key = sTopKey[tid];
    unsigned u = (key & 0x80000000u) ? (key & 0x7FFFFFFFu) : ~key;
    float val = __uint_as_float(u);
    bool valid = (val > CONF_THR);
    unsigned idx = sTopIdx[tid];
    size_t o = (bn + (size_t)idx) * 4;
    cx = locWS[o]; cy = locWS[o + 1]; w = locWS[o + 2]; h = locWS[o + 3];
    float x1 = cx - 0.5f * w, y1 = cy - 0.5f * h;
    float x2 = cx + 0.5f * w, y2 = cy + 0.5f * h;
    sX1[tid] = x1; sY1[tid] = y1; sX2[tid] = x2; sY2[tid] = y2;
    sAr[tid] = (x2 - x1) * (y2 - y1);
    sKeep[tid] = valid ? 1 : 0;
  } else {
    sKeep[tid] = 0;
  }
  __syncthreads();

  // Greedy NMS: 200 sequential rounds, IoU computed on the fly.
  for (int i = 0; i < TOPK; ++i) {
    bool keepI = (sKeep[i] != 0);
    float ix1 = sX1[i], iy1 = sY1[i], ix2 = sX2[i], iy2 = sY2[i], iar = sAr[i];
    if (keepI && tid > i && tid < TOPK && sKeep[tid]) {
      float iw = fminf(ix2, sX2[tid]) - fmaxf(ix1, sX1[tid]);
      float ih = fminf(iy2, sY2[tid]) - fmaxf(iy1, sY1[tid]);
      iw = fmaxf(iw, 0.0f); ih = fmaxf(ih, 0.0f);
      float inter = iw * ih;
      float iou = inter / (iar + sAr[tid] - inter);
      if (iou > IOU_THR) sKeep[tid] = 0;
    }
    __syncthreads();
  }

  // Emit [label, cx, cy, w, h] and keep flag.
  if (tid < TOPK) {
    size_t ob = ((size_t)b * NCLS + c) * TOPK + tid;
    float* o5 = outBoxes + ob * 5;
    o5[0] = (float)c;
    o5[1] = cx; o5[2] = cy; o5[3] = w; o5[4] = h;
    outKeep[ob] = sKeep[tid] ? 1.0f : 0.0f;
  }
}

extern "C" void kernel_launch(void* const* d_in, const int* in_sizes, int n_in,
                              void* d_out, int out_size, void* d_ws,
                              size_t ws_size, hipStream_t stream) {
  const float* predicts = (const float*)d_in[0];
  const float* dboxes   = (const float*)d_in[1];

  float* outBoxes = (float*)d_out;                                   // (B,80,200,5)
  float* outKeep  = outBoxes + (size_t)BATCH * NCLS * TOPK * 5;      // (B,80,200)

  float* locWS    = (float*)d_ws;                                    // B*N*4
  float* rowMaxWS = locWS + (size_t)BATCH * NBOX * 4;                // B*N
  float* invSumWS = rowMaxWS + (size_t)BATCH * NBOX;                 // B*N

  dim3 g1((NBOX + 127) / 128, BATCH);
  phase1_stats_decode<<<g1, 256, 0, stream>>>(predicts, dboxes, locWS,
                                              rowMaxWS, invSumWS);

  // Dynamic LDS: keys + hist + top lists + box arrays + control.
  size_t shDwords = (size_t)NPAD + 256 + 256 + 256 + 5 * 256 + 256 + 8 + 8 + 4;
  dim3 g2(NCLS, BATCH);  // class fastest -> same-batch classes co-resident in L2
  phase2_topk_nms<<<g2, 256, shDwords * 4, stream>>>(
      predicts, locWS, rowMaxWS, invSumWS, outBoxes, outKeep);
}
